// STABSA_12687333392639
// MI455X (gfx1250) — matrix operand order repaired and checked by reading, not compiled, not run on hardware
//
#include <hip/hip_runtime.h>

typedef _Float16 half_t;
typedef __attribute__((ext_vector_type(16))) _Float16 v16h;
typedef __attribute__((ext_vector_type(8)))  _Float16 v8h;
typedef __attribute__((ext_vector_type(8)))  float    v8f;

#define BB   4
#define LS   160
#define LW   128
#define DD   768
#define NSPAN 996
#define NZ   64
#define NPAIR 4096
#define HID  128
#define AFK  832     // 768 feat + 64 pos

// ---------- CDNA5 async global->LDS copy (16B per lane), ASYNCcnt tracked ----------
__device__ __forceinline__ void async_copy_b128(unsigned int ldsAddr, const void* gaddr) {
    asm volatile("global_load_async_to_lds_b128 %0, %1, off"
                 :: "v"(ldsAddr), "v"(gaddr) : "memory");
}
__device__ __forceinline__ void wait_async0() {
    asm volatile("s_wait_asynccnt 0" ::: "memory");
}
__device__ __forceinline__ unsigned int lds_addr_of(const void* p) {
    return (unsigned int)(unsigned long long)p;   // low 32 bits of shared aperture = LDS offset
}

// ---------- small device helpers ----------
__device__ __forceinline__ int bucket17(int x) {
    const int bins[17] = {0,1,2,3,4,5,6,7,8,9,10,15,20,25,30,50,80};
    int id = 0;
    #pragma unroll
    for (int i = 1; i < 17; ++i) if (bins[i] <= x) id = i;
    return id;
}
__device__ __forceinline__ void decode_span(int s, int& st, int& en, int& w) {
    int rem = s, ww = 1;
    while (rem >= (LW + 1 - ww)) { rem -= (LW + 1 - ww); ++ww; }
    st = rem; en = rem + ww - 1; w = ww;
}

// ---------- batched f16 WMMA GEMM with async-LDS staging ----------
// C[b] = act(A[b](f16)[M,K] @ Wt[b](f16)[N,K]^T + bias)
// block = 256 threads (8 waves), tile 32(M) x 64(N), K step 32. K%32==0, N%64==0.
__global__ __launch_bounds__(256) void k_gemm_a(
    const half_t* __restrict__ A, int lda, long strideA, int M,
    const half_t* __restrict__ Wt, int ldw, long strideW, int N, int K,
    const float* __restrict__ bias, int relu,
    float* __restrict__ outF32, half_t* __restrict__ outF16, int ldc, long strideC)
{
    __shared__ __attribute__((aligned(32))) half_t As[32*32];
    __shared__ __attribute__((aligned(32))) half_t Bs[64*32];
    const int t = threadIdx.x, lane = t & 31, wave = t >> 5;
    const int mrow = (wave >> 2) << 4;      // 0 / 16
    const int ncol = (wave & 3) << 4;       // 0..48
    const int rowBase = blockIdx.y * 32;
    const int colBase = blockIdx.x * 64;
    const int bz = blockIdx.z;
    const half_t* Ab = A  + (long)bz * strideA;
    const half_t* Wb = Wt + (long)bz * strideW;

    const int aRow = t >> 2, aK = (t & 3) << 3;   // A: threads 0..127 -> 32x32 tile in 8-half chunks
    const int bCol = t >> 2, bK = (t & 3) << 3;   // B: 256 threads -> 64x32 tile in 8-half chunks
    const unsigned int asDst = lds_addr_of(&As[aRow*32 + aK]);
    const unsigned int bsDst = lds_addr_of(&Bs[bCol*32 + bK]);

    v8f acc = {};
    for (int kb = 0; kb < K; kb += 32) {
        if (t < 128)
            async_copy_b128(asDst, Ab + (long)(rowBase + aRow)*lda + kb + aK);
        async_copy_b128(bsDst, Wb + (long)(colBase + bCol)*ldw + kb + bK);
        wait_async0();
        __syncthreads();
        // A fragment: lane<16 -> K 0..15, lane>=16 -> K 16..31
        int arow = mrow + (lane & 15);
        int aoff = arow*32 + ((lane >> 4) << 4);
        v8h a0 = *(const v8h*)(&As[aoff]);
        v8h a1 = *(const v8h*)(&As[aoff + 8]);
        v16h af;
        #pragma unroll
        for (int i = 0; i < 8; ++i) { af[i] = a0[i]; af[i+8] = a1[i]; }
        // B fragment: lane<16 -> K 0..15, lane>=16 -> K 16..31 (contiguous per column)
        v16h bf = *(const v16h*)(&Bs[(ncol + (lane & 15))*32 + ((lane >> 4) << 4)]);
        acc = __builtin_amdgcn_wmma_f32_16x16x32_f16(false, af, false, bf, (short)0, acc, false, false);
        __syncthreads();
    }
    int col = colBase + ncol + (lane & 15);
    float bv = bias ? bias[col] : 0.f;
    #pragma unroll
    for (int r = 0; r < 8; ++r) {
        int row = rowBase + mrow + r + ((lane >> 4) << 3);
        if (row < M) {
            float v = acc[r] + bv;
            if (relu) v = v > 0.f ? v : 0.f;
            if (outF32) outF32[(long)bz*strideC + (long)row*ldc + col] = v;
            if (outF16) outF16[(long)bz*strideC + (long)row*ldc + col] = (half_t)v;
        }
    }
}

// ---------- operand preparation (one-time f16 transposed copies) ----------

// Amask[b*LW+w][k] = mask[b, fs[b,w], k] / wlen[b,w]   (fold word_len into A rows)
__global__ __launch_bounds__(256) void k_amask(
    const float* __restrict__ mask, const int* __restrict__ fs,
    const float* __restrict__ wlen, half_t* __restrict__ Amask)
{
    int tid = blockIdx.x*256 + threadIdx.x;           // B*LW*LS
    int k = tid % LS, row = tid / LS;                  // row = b*LW + w
    int b = row >> 7;
    int fsRow = fs[row];
    float inv = 1.0f / wlen[row];
    Amask[(long)row*LS + k] = (half_t)(mask[((long)b*LS + fsRow)*LS + k] * inv);
}

// hiddenT[b][n][k] = hidden[b][k][n]  (f16, per-batch [768][160])
__global__ __launch_bounds__(256) void k_hiddenT(
    const float* __restrict__ hidden, half_t* __restrict__ hiddenT)
{
    int tid = blockIdx.x*256 + threadIdx.x;           // B*DD*LS
    int k = tid % LS, rem = tid / LS;
    int n = rem % DD, b = rem / DD;
    hiddenT[tid] = (half_t)hidden[((long)b*LS + k)*DD + n];
}

// WpackT[768][832] f16: six projection column-blocks, transposed [N][K]
__global__ __launch_bounds__(256) void k_pack(
    const float* __restrict__ sW1, const float* __restrict__ pW1,
    half_t* __restrict__ WpackT)
{
    const int featBase[6] = {0, 768, 0, 768, 1728, 2496};
    const int posBase[6]  = {1536, 1600, 1536, 1600, 3264, 3328};
    int tid = blockIdx.x*256 + threadIdx.x;           // 768*832
    int r = tid % AFK, n = tid / AFK;                  // n = j*128 + c
    int j = n >> 7, c = n & 127;
    const float* src = (j < 2) ? sW1 : pW1;
    int srcRow = (r < 768) ? (featBase[j] + r) : (posBase[j] + (r - 768));
    WpackT[tid] = (half_t)src[srcRow*HID + c];
}

// sW2T / pW2T f16 [128][128] transposed
__global__ __launch_bounds__(256) void k_w2t(
    const float* __restrict__ sW2, const float* __restrict__ pW2,
    half_t* __restrict__ sW2T, half_t* __restrict__ pW2T)
{
    int tid = blockIdx.x*256 + threadIdx.x;           // 2*128*128
    int k = tid & 127, n = (tid >> 7) & 127, w = tid >> 14;
    if (w == 0) sW2T[n*HID + k] = (half_t)sW2[k*HID + n];
    else        pW2T[n*HID + k] = (half_t)pW2[k*HID + n];
}

// ---------- Afeat pos columns ----------
__global__ __launch_bounds__(256) void k_posfill(
    const float* __restrict__ pos_table, const int* __restrict__ pos_id,
    half_t* __restrict__ Afeat)
{
    int tid = blockIdx.x*256 + threadIdx.x;           // B*LW*64
    int row = tid >> 6, k = tid & 63;
    Afeat[(long)row*AFK + 768 + k] = (half_t)pos_table[pos_id[row]*64 + k];
}

// ---------- tiny table GEMMs: SmallTab[98][128] ----------
// rows 0..16 WW_s | 17..33 WA | 34..50 WB | 51..67 RelW | 68..97 DepW
__global__ __launch_bounds__(256) void k_tables(
    const float* __restrict__ width_table, const float* __restrict__ rel_table,
    const float* __restrict__ dep_table,
    const float* __restrict__ sW1, const float* __restrict__ pW1,
    float* __restrict__ SmallTab)
{
    int tid = blockIdx.x*256 + threadIdx.x;           // 98*128
    int row = tid >> 7, col = tid & 127;
    const float* emb; const float* Wsrc; int base;
    if      (row < 17) { emb = width_table + (row)*64;    Wsrc = sW1; base = 1664; }
    else if (row < 34) { emb = width_table + (row-17)*64; Wsrc = pW1; base = 1664; }
    else if (row < 51) { emb = width_table + (row-34)*64; Wsrc = pW1; base = 3392; }
    else if (row < 68) { emb = rel_table   + (row-51)*64; Wsrc = pW1; base = 3456; }
    else               { emb = dep_table   + (row-68)*64; Wsrc = pW1; base = 3520; }
    float acc = 0.f;
    #pragma unroll 8
    for (int k = 0; k < 64; ++k) acc += emb[k] * Wsrc[(base + k)*HID + col];
    SmallTab[row*HID + col] = acc;
}

// ---------- span hidden-1 assembly (factorized concat-GEMM) ----------
__global__ __launch_bounds__(256) void k_span_h1(
    const float* __restrict__ Proj, const float* __restrict__ SmallTab,
    const float* __restrict__ sb1, half_t* __restrict__ SpanH1)
{
    int tid = blockIdx.x*256 + threadIdx.x;           // 3984*128
    int row = tid >> 7, col = tid & 127;
    int b = row / NSPAN, s = row - b*NSPAN;
    int st, en, w; decode_span(s, st, en, w);
    int wid = bucket17(w);
    float v = Proj[(long)(b*LW + st)*768 + col]
            + Proj[(long)(b*LW + en)*768 + 128 + col]
            + SmallTab[wid*HID + col] + sb1[col];
    SpanH1[(long)row*HID + col] = (half_t)(v > 0.f ? v : 0.f);
}

// ---------- span layer-3 (N=3, scalar) ----------
__global__ __launch_bounds__(256) void k_span_out(
    const half_t* __restrict__ H2, const float* __restrict__ sW3,
    const float* __restrict__ sb3, float* __restrict__ out)
{
    int row = blockIdx.x*256 + threadIdx.x;
    if (row >= BB*NSPAN) return;
    float a0 = sb3[0], a1 = sb3[1], a2 = sb3[2];
    const half_t* h = H2 + (long)row*HID;
    for (int k = 0; k < HID; ++k) {
        float x = (float)h[k];
        a0 += x * sW3[k*3+0]; a1 += x * sW3[k*3+1]; a2 += x * sW3[k*3+2];
    }
    out[row*3+0] = a0; out[row*3+1] = a1; out[row*3+2] = a2;
}

// ---------- top-k (iterative argmax, jax tie-breaking) ----------
__global__ __launch_bounds__(256) void k_topk(
    const float* __restrict__ spanProb,
    int* __restrict__ selS, int* __restrict__ selE, int* __restrict__ selW)
{
    __shared__ float sv[NSPAN];
    __shared__ float bv[256];
    __shared__ int   bi[256];
    int b = blockIdx.x >> 1, cls = blockIdx.x & 1, t = threadIdx.x;
    for (int s = t; s < NSPAN; s += 256) sv[s] = spanProb[(b*NSPAN + s)*3 + 1 + cls];
    __syncthreads();
    for (int it = 0; it < NZ; ++it) {
        float best = -3.4e38f; int bestI = NSPAN;
        for (int s = t; s < NSPAN; s += 256) {
            float v = sv[s];
            if (v > best || (v == best && s < bestI)) { best = v; bestI = s; }
        }
        bv[t] = best; bi[t] = bestI;
        __syncthreads();
        for (int off = 128; off; off >>= 1) {
            if (t < off) {
                float v2 = bv[t+off]; int i2 = bi[t+off];
                if (v2 > bv[t] || (v2 == bv[t] && i2 < bi[t])) { bv[t] = v2; bi[t] = i2; }
            }
            __syncthreads();
        }
        if (t == 0) {
            int sel = bi[0];
            sv[sel] = -3.4e38f;
            int st, en, w; decode_span(sel, st, en, w);
            int o = (b*2 + cls)*NZ + it;
            selS[o] = st; selE[o] = en; selW[o] = bucket17(w);
        }
        __syncthreads();
    }
}

// ---------- U_t / U_o : projected selected-span rows ----------
__global__ __launch_bounds__(256) void k_U(
    const float* __restrict__ Proj, const float* __restrict__ SmallTab,
    const int* __restrict__ selS, const int* __restrict__ selE, const int* __restrict__ selW,
    float* __restrict__ U)
{
    int tid = blockIdx.x*256 + threadIdx.x;           // 4*2*64*128
    int col = tid & 127, i = (tid >> 7) & 63, side = (tid >> 13) & 1, b = tid >> 14;
    int o = (b*2 + side)*NZ + i;
    int st = selS[o], en = selE[o], wid = selW[o];
    int c1 = 256 + side*256, c2 = 384 + side*256;
    int tabRow = (side ? 34 : 17) + wid;
    U[(long)o*HID + col] = Proj[(long)(b*LW + st)*768 + c1 + col]
                         + Proj[(long)(b*LW + en)*768 + c2 + col]
                         + SmallTab[tabRow*HID + col];
}

// ---------- per-pair metadata: rel bucket + dep block-min ----------
__global__ __launch_bounds__(256) void k_pairmeta(
    const int* __restrict__ selS, const int* __restrict__ selE,
    const int* __restrict__ dep_dis, int* __restrict__ relId, int* __restrict__ depId)
{
    int tid = blockIdx.x*256 + threadIdx.x;           // 4*4096
    int b = tid >> 12, p = tid & 4095, i = p >> 6, j = p & 63;
    int a  = selS[(b*2+0)*NZ + i], bb = selE[(b*2+0)*NZ + i];
    int c  = selS[(b*2+1)*NZ + j], d  = selE[(b*2+1)*NZ + j];
    int rel = min(abs(bb - c), abs(a - d));
    int m = 0x7fffffff;
    for (int r = a; r <= bb; ++r)
        for (int cc = c; cc <= d; ++cc)
            m = min(m, dep_dis[((long)b*LW + r)*LW + cc]);
    relId[tid] = bucket17(rel);
    depId[tid] = m;                                   // raw min distance indexes dep_table
}

// ---------- pair hidden-1 assembly ----------
__global__ __launch_bounds__(256) void k_pair_h1(
    const float* __restrict__ U, const float* __restrict__ SmallTab,
    const int* __restrict__ relId, const int* __restrict__ depId,
    const float* __restrict__ pb1, half_t* __restrict__ PairH1)
{
    int tid = blockIdx.x*256 + threadIdx.x;           // 16384*128
    int row = tid >> 7, col = tid & 127;
    int b = row >> 12, p = row & 4095, i = p >> 6, j = p & 63;
    float v = U[(long)((b*2+0)*NZ + i)*HID + col]
            + U[(long)((b*2+1)*NZ + j)*HID + col]
            + SmallTab[(51 + relId[row])*HID + col]
            + SmallTab[(68 + depId[row])*HID + col]
            + pb1[col];
    PairH1[(long)row*HID + col] = (half_t)(v > 0.f ? v : 0.f);
}

// ---------- pair layer-3 (N=4, scalar) ----------
__global__ __launch_bounds__(256) void k_pair_out(
    const half_t* __restrict__ H2, const float* __restrict__ pW3,
    const float* __restrict__ pb3, float* __restrict__ out)
{
    int row = blockIdx.x*256 + threadIdx.x;           // 16384
    float a0 = pb3[0], a1 = pb3[1], a2 = pb3[2], a3 = pb3[3];
    const half_t* h = H2 + (long)row*HID;
    for (int k = 0; k < HID; ++k) {
        float x = (float)h[k];
        a0 += x * pW3[k*4+0]; a1 += x * pW3[k*4+1];
        a2 += x * pW3[k*4+2]; a3 += x * pW3[k*4+3];
    }
    out[row*4+0] = a0; out[row*4+1] = a1; out[row*4+2] = a2; out[row*4+3] = a3;
}

extern "C" void kernel_launch(void* const* d_in, const int* in_sizes, int n_in,
                              void* d_out, int out_size, void* d_ws, size_t ws_size,
                              hipStream_t stream) {
    (void)in_sizes; (void)n_in; (void)out_size; (void)ws_size;
    const float* last_hidden = (const float*)d_in[0];
    const float* mask        = (const float*)d_in[1];
    const float* word_len    = (const float*)d_in[2];
    const float* pos_table   = (const float*)d_in[3];
    const float* width_table = (const float*)d_in[4];
    const float* rel_table   = (const float*)d_in[5];
    const float* dep_table   = (const float*)d_in[6];
    const float* sW1 = (const float*)d_in[7];  const float* sb1 = (const float*)d_in[8];
    const float* sW2 = (const float*)d_in[9];  const float* sb2 = (const float*)d_in[10];
    const float* sW3 = (const float*)d_in[11]; const float* sb3 = (const float*)d_in[12];
    const float* pW1 = (const float*)d_in[13]; const float* pb1 = (const float*)d_in[14];
    const float* pW2 = (const float*)d_in[15]; const float* pb2 = (const float*)d_in[16];
    const float* pW3 = (const float*)d_in[17]; const float* pb3 = (const float*)d_in[18];
    const int* first_subword = (const int*)d_in[19];
    const int* pos_id        = (const int*)d_in[20];
    const int* dep_dis       = (const int*)d_in[21];
    float* out = (float*)d_out;
    float* span_out = out;                      // [4*996*3]
    float* pair_out = out + BB*NSPAN*3;         // [4*4096*4]

    char* ws = (char*)d_ws; size_t off = 0;
    auto alloc = [&](size_t bytes) -> void* {
        void* p = ws + off; off = (off + bytes + 255) & ~(size_t)255; return p;
    };
    half_t* WpackT   = (half_t*)alloc((size_t)768*AFK*2);      // [N=768][K=832]
    half_t* sW2T     = (half_t*)alloc((size_t)HID*HID*2);
    half_t* pW2T     = (half_t*)alloc((size_t)HID*HID*2);
    half_t* hiddenT  = (half_t*)alloc((size_t)BB*DD*LS*2);     // [B][768][160]
    half_t* Amask    = (half_t*)alloc((size_t)BB*LW*LS*2);     // [B*128][160]
    float*  SmallTab = (float*) alloc(98*HID*4);
    half_t* Afeat    = (half_t*)alloc((size_t)BB*LW*AFK*2);    // [512][832]
    float*  Proj     = (float*) alloc((size_t)BB*LW*768*4);
    half_t* SpanH1   = (half_t*)alloc((size_t)(BB*NSPAN + 16)*HID*2);  // +16 rows tile pad
    half_t* H2s      = (half_t*)alloc((size_t)BB*NSPAN*HID*2);
    int*    selS     = (int*)   alloc(BB*2*NZ*4);
    int*    selE     = (int*)   alloc(BB*2*NZ*4);
    int*    selW     = (int*)   alloc(BB*2*NZ*4);
    float*  U        = (float*) alloc((size_t)BB*2*NZ*HID*4);
    int*    relId    = (int*)   alloc(BB*NPAIR*4);
    int*    depId    = (int*)   alloc(BB*NPAIR*4);
    half_t* PairH1   = (half_t*)alloc((size_t)BB*NPAIR*HID*2);
    half_t* H2p      = (half_t*)alloc((size_t)BB*NPAIR*HID*2);

    // 1) one-time f16/transposed operand prep + small tables
    k_pack<<<(768*AFK)/256, 256, 0, stream>>>(sW1, pW1, WpackT);
    k_w2t<<<(2*HID*HID)/256, 256, 0, stream>>>(sW2, pW2, sW2T, pW2T);
    k_hiddenT<<<(BB*DD*LS)/256, 256, 0, stream>>>(last_hidden, hiddenT);
    k_amask<<<(BB*LW*LS)/256, 256, 0, stream>>>(mask, first_subword, word_len, Amask);
    k_tables<<<(98*HID)/256, 256, 0, stream>>>(width_table, rel_table, dep_table, sW1, pW1, SmallTab);

    // 2) word features (batched WMMA GEMM): Afeat[:, :768] = Amask @ hiddenT^T
    k_gemm_a<<<dim3(DD/64, LW/32, BB), 256, 0, stream>>>(
        Amask, LS, (long)LW*LS, LW, hiddenT, LS, (long)DD*LS, DD, LS,
        nullptr, 0, nullptr, Afeat, AFK, (long)LW*AFK);
    k_posfill<<<(BB*LW*64)/256, 256, 0, stream>>>(pos_table, pos_id, Afeat);

    // 3) all six word projections in one WMMA GEMM: [512,832]@[832,768]
    k_gemm_a<<<dim3(768/64, (BB*LW)/32, 1), 256, 0, stream>>>(
        Afeat, AFK, 0, BB*LW, WpackT, AFK, 0, 768, AFK,
        nullptr, 0, Proj, nullptr, 768, 0);

    // 4) span branch
    k_span_h1<<<(BB*NSPAN*HID)/256, 256, 0, stream>>>(Proj, SmallTab, sb1, SpanH1);
    k_gemm_a<<<dim3(HID/64, (BB*NSPAN + 31)/32, 1), 256, 0, stream>>>(
        SpanH1, HID, 0, BB*NSPAN, sW2T, HID, 0, HID, HID,
        sb2, 1, nullptr, H2s, HID, 0);
    k_span_out<<<(BB*NSPAN + 255)/256, 256, 0, stream>>>(H2s, sW3, sb3, span_out);

    // 5) top-k pruning (per batch, per class)
    k_topk<<<BB*2, 256, 0, stream>>>(span_out, selS, selE, selW);

    // 6) pair branch
    k_U<<<(BB*2*NZ*HID)/256, 256, 0, stream>>>(Proj, SmallTab, selS, selE, selW, U);
    k_pairmeta<<<(BB*NPAIR)/256, 256, 0, stream>>>(selS, selE, dep_dis, relId, depId);
    k_pair_h1<<<((long)BB*NPAIR*HID)/256, 256, 0, stream>>>(U, SmallTab, relId, depId, pb1, PairH1);
    k_gemm_a<<<dim3(HID/64, (BB*NPAIR)/32, 1), 256, 0, stream>>>(
        PairH1, HID, 0, BB*NPAIR, pW2T, HID, 0, HID, HID,
        pb2, 1, nullptr, H2p, HID, 0);
    k_pair_out<<<(BB*NPAIR)/256, 256, 0, stream>>>(H2p, pW3, pb3, pair_out);
}
